// PointConvSceneFlowPWC8192_1417339207993
// MI455X (gfx1250) — matrix operand order repaired and checked by reading, not compiled, not run on hardware
//
#include <hip/hip_runtime.h>
#include <stddef.h>
#include <stdint.h>

// =====================================================================
// PointPWC (PointConvSceneFlowPWC8192) forward for MI455X / gfx1250.
// - GEMMs: v_wmma_f32_16x16x32_bf16, 16x64 tile/wave (A reused over 4 WMMAs),
//   branchless clamped fragment loads (float4 fast path), f32 accumulate,
//   wave-uniform full-tile store fast path (no divergent store predication).
// - FPS: register-resident points, one 1024-thread workgroup.
// - KNN: VGPR-resident sorted top-k lists, LDS-tiled reference streaming.
// - Whole activation set < 192MB -> L2-resident; single-stream reuse.
// =====================================================================

typedef __attribute__((ext_vector_type(16))) __bf16 v16bf;
typedef __attribute__((ext_vector_type(8)))  float  v8f;

static __device__ __forceinline__ float leakyf(float x) { return x >= 0.f ? x : 0.1f * x; }
static __device__ __forceinline__ float reluf(float x)  { return x > 0.f ? x : 0.f; }

static __device__ __forceinline__ float actf(float r, int act) {
    if (act == 1)      r = leakyf(r);
    else if (act == 3) r = fminf(fmaxf(r, -200.f), 200.f);
    return r;
}

template <int B0>
static __device__ __forceinline__ void pk8(v16bf& a, float4 u0, float4 u1) {
    a[B0 + 0] = (__bf16)u0.x; a[B0 + 1] = (__bf16)u0.y;
    a[B0 + 2] = (__bf16)u0.z; a[B0 + 3] = (__bf16)u0.w;
    a[B0 + 4] = (__bf16)u1.x; a[B0 + 5] = (__bf16)u1.y;
    a[B0 + 6] = (__bf16)u1.z; a[B0 + 7] = (__bf16)u1.w;
}

// --------------------------------------------------------------------
// WMMA GEMM: Y[N,Co] = act(X[N,Ci] @ W[Co,Ci]^T + b)
// act: 0 none, 1 leaky(0.1), 3 clamp(+-200)
// 16(M) x 64(Co) tile per wave: one A fragment feeds 4 WMMAs.
// Out-of-range rows/cols are CLAMPED in the K loop (their D rows/cols
// are never stored). Stores use a wave-uniform full-tile fast path.
// --------------------------------------------------------------------
template <bool VEC>
__global__ __launch_bounds__(256)
void k_gemm(const float* __restrict__ X, const float* __restrict__ W,
            const float* __restrict__ Bp, float* __restrict__ Y,
            int N, int Ci, int Co, int act)
{
    const int lane = threadIdx.x & 31;
    const int wave = blockIdx.x * (blockDim.x >> 5) + (threadIdx.x >> 5);
    const int tilesCo = (Co + 63) >> 6;
    const int tilesN  = (N + 15) >> 4;
    if (wave >= tilesCo * tilesN) return;          // whole-wave uniform
    const int m0 = (wave / tilesCo) << 4;
    const int n0 = (wave % tilesCo) << 6;
    const int half = lane >> 4, l = lane & 15;
    const int arow = min(m0 + l, N - 1);           // clamped: row >= N never stored
    const float* __restrict__ xr = X + (size_t)arow * Ci;
    const int bc0 = n0 + l;
    const float* __restrict__ wr0 = W + (size_t)min(bc0,      Co - 1) * Ci;
    const float* __restrict__ wr1 = W + (size_t)min(bc0 + 16, Co - 1) * Ci;
    const float* __restrict__ wr2 = W + (size_t)min(bc0 + 32, Co - 1) * Ci;
    const float* __restrict__ wr3 = W + (size_t)min(bc0 + 48, Co - 1) * Ci;

    v8f acc0 = {}, acc1 = {}, acc2 = {}, acc3 = {};
    const int Civ = Ci & ~31;
    int kk = 0;
    for (; kk < Civ; kk += 32) {                   // unconditional main loop
        v16bf a, b0, b1, b2, b3;
        const int kA0 = kk + half * 16;            // A: half0 K kk+0..7 / kk+8..15
        const int kA1 = kk + half * 16 + 8;        //    half1 K kk+16..23 / kk+24..31
        const int kB  = kk + half * 16;            // B: half0 K kk+0..15, half1 kk+16..31
        if (VEC) {
            pk8<0>(a, *(const float4*)(xr + kA0), *(const float4*)(xr + kA0 + 4));
            pk8<8>(a, *(const float4*)(xr + kA1), *(const float4*)(xr + kA1 + 4));
            pk8<0>(b0, *(const float4*)(wr0 + kB),     *(const float4*)(wr0 + kB + 4));
            pk8<8>(b0, *(const float4*)(wr0 + kB + 8), *(const float4*)(wr0 + kB + 12));
            pk8<0>(b1, *(const float4*)(wr1 + kB),     *(const float4*)(wr1 + kB + 4));
            pk8<8>(b1, *(const float4*)(wr1 + kB + 8), *(const float4*)(wr1 + kB + 12));
            pk8<0>(b2, *(const float4*)(wr2 + kB),     *(const float4*)(wr2 + kB + 4));
            pk8<8>(b2, *(const float4*)(wr2 + kB + 8), *(const float4*)(wr2 + kB + 12));
            pk8<0>(b3, *(const float4*)(wr3 + kB),     *(const float4*)(wr3 + kB + 4));
            pk8<8>(b3, *(const float4*)(wr3 + kB + 8), *(const float4*)(wr3 + kB + 12));
        } else {
#pragma unroll
            for (int e = 0; e < 8; ++e) {
                a[e]     = (__bf16)xr[kA0 + e];
                a[e + 8] = (__bf16)xr[kA1 + e];
            }
#pragma unroll
            for (int e = 0; e < 16; ++e) {
                b0[e] = (__bf16)wr0[kB + e];
                b1[e] = (__bf16)wr1[kB + e];
                b2[e] = (__bf16)wr2[kB + e];
                b3[e] = (__bf16)wr3[kB + e];
            }
        }
        acc0 = __builtin_amdgcn_wmma_f32_16x16x32_bf16(false, a, false, b0, (short)0, acc0, false, false);
        acc1 = __builtin_amdgcn_wmma_f32_16x16x32_bf16(false, a, false, b1, (short)0, acc1, false, false);
        acc2 = __builtin_amdgcn_wmma_f32_16x16x32_bf16(false, a, false, b2, (short)0, acc2, false, false);
        acc3 = __builtin_amdgcn_wmma_f32_16x16x32_bf16(false, a, false, b3, (short)0, acc3, false, false);
    }
    if (kk < Ci) {                                 // single tail step: clamp + cndmask
        v16bf a, b0, b1, b2, b3;
        const int kA0 = kk + half * 16;
        const int kA1 = kk + half * 16 + 8;
        const int kB  = kk + half * 16;
#pragma unroll
        for (int e = 0; e < 8; ++e) {
            const int k0 = kA0 + e, k1 = kA1 + e;
            const float f0 = xr[min(k0, Ci - 1)];
            const float f1 = xr[min(k1, Ci - 1)];
            a[e]     = (__bf16)(k0 < Ci ? f0 : 0.f);
            a[e + 8] = (__bf16)(k1 < Ci ? f1 : 0.f);
        }
#pragma unroll
        for (int e = 0; e < 16; ++e) {
            const int k = kB + e;
            const int kc = min(k, Ci - 1);
            const bool ok = k < Ci;
            b0[e] = (__bf16)(ok ? wr0[kc] : 0.f);
            b1[e] = (__bf16)(ok ? wr1[kc] : 0.f);
            b2[e] = (__bf16)(ok ? wr2[kc] : 0.f);
            b3[e] = (__bf16)(ok ? wr3[kc] : 0.f);
        }
        acc0 = __builtin_amdgcn_wmma_f32_16x16x32_bf16(false, a, false, b0, (short)0, acc0, false, false);
        acc1 = __builtin_amdgcn_wmma_f32_16x16x32_bf16(false, a, false, b1, (short)0, acc1, false, false);
        acc2 = __builtin_amdgcn_wmma_f32_16x16x32_bf16(false, a, false, b2, (short)0, acc2, false, false);
        acc3 = __builtin_amdgcn_wmma_f32_16x16x32_bf16(false, a, false, b3, (short)0, acc3, false, false);
    }
    const bool rowFull = (m0 + 16) <= N;           // wave-uniform
#pragma unroll
    for (int t = 0; t < 4; ++t) {                  // D: VGPR v -> row v (+8 upper half)
        const int bc = n0 + t * 16 + l;
        const v8f accv = (t == 0) ? acc0 : (t == 1) ? acc1 : (t == 2) ? acc2 : acc3;
        const bool colFull = (n0 + t * 16 + 16) <= Co;   // wave-uniform
        if (colFull && rowFull) {
            // fast path: no predication at all
            const float bias = Bp ? Bp[bc] : 0.f;
            float* __restrict__ yp = Y + (size_t)(m0 + half * 8) * Co + bc;
#pragma unroll
            for (int v = 0; v < 8; ++v)
                yp[(size_t)v * Co] = actf(accv[v] + bias, act);
        } else if (bc < Co) {
            const float bias = Bp ? Bp[bc] : 0.f;
#pragma unroll
            for (int v = 0; v < 8; ++v) {
                const int row = m0 + v + half * 8;
                if (row < N)
                    Y[(size_t)row * Co + bc] = actf(accv[v] + bias, act);
            }
        }
    }
}

// --------------------------------------------------------------------
// Per-point einsum: out[n, c, w] = sum_k pts[n,k,c] * wts[n,k,w]
// (k = KK <= 16, w-dim fixed 16). One wave per (n, 16-channel tile),
// single zero-padded bf16 WMMA. Branchless clamped loads; the K>=16
// half of the fragments is statically zero since KK <= 16.
// --------------------------------------------------------------------
__global__ __launch_bounds__(256)
void k_pc_einsum(const float* __restrict__ pts, const float* __restrict__ wts,
                 float* __restrict__ out, int N, int KK, int C)
{
    const int lane = threadIdx.x & 31;
    const int wave = blockIdx.x * (blockDim.x >> 5) + (threadIdx.x >> 5);
    const int tilesC = (C + 15) >> 4;
    if (wave >= N * tilesC) return;
    const int n  = wave / tilesC;
    const int c0 = (wave % tilesC) << 4;
    const int half = lane >> 4, l = lane & 15;
    const int mi = min(c0 + l, C - 1);             // clamped: row >= C never stored
    const float* __restrict__ Pn = pts + (size_t)n * KK * C;
    const float* __restrict__ Wn = wts + (size_t)n * KK * 16;
    v16bf a, b;
#pragma unroll
    for (int e = 0; e < 8; ++e) {
        const int k0 = half * 8 + e;               // 0..15
        const float f = Pn[(size_t)min(k0, KK - 1) * C + mi];
        a[e]     = (__bf16)(k0 < KK ? f : 0.f);
        a[e + 8] = (__bf16)0.f;                    // K 16..31 always past KK
    }
#pragma unroll
    for (int e = 0; e < 16; ++e) {
        const int k = (e < 8) ? half * 8 + e : 16 + half * 8 + (e - 8);   // e >= 8 -> zero (k >= 16)
        const float f = Wn[min(k, KK - 1) * 16 + l];
        b[e] = (__bf16)(k < KK ? f : 0.f);
    }
    v8f acc = {};
    acc = __builtin_amdgcn_wmma_f32_16x16x32_bf16(false, a, false, b,
                                                  (short)0, acc, false, false);
    float* __restrict__ op = out + (size_t)n * C * 16 + (size_t)(c0 + half * 8) * 16 + l;
    if (c0 + 16 <= C) {                            // wave-uniform fast path
#pragma unroll
        for (int v = 0; v < 8; ++v) op[(size_t)v * 16] = acc[v];
    } else {
#pragma unroll
        for (int v = 0; v < 8; ++v) {
            const int row = c0 + v + half * 8;
            if (row < C) out[(size_t)n * C * 16 + (size_t)row * 16 + l] = acc[v];
        }
    }
}

// --------------------------------------------------------------------
// Farthest point sampling, seeded at index 0. One 1024-thread WG.
// Points + running min-distances live in VGPRs (<=8 per thread);
// LDS used only for the argmax reduction and farthest-point broadcast.
// --------------------------------------------------------------------
#define FPS_TPB 1024
__global__ __launch_bounds__(FPS_TPB)
void k_fps(const float* __restrict__ xyz, int N, int npoint, int* __restrict__ outIdx)
{
    __shared__ float s_val[FPS_TPB];
    __shared__ int   s_idx[FPS_TPB];
    __shared__ float s_far[3];
    const int tid = threadIdx.x;
    float px[8], py[8], pz[8], pd[8];
    int cnt = 0;
    for (int i = tid; i < N; i += FPS_TPB) {
        px[cnt] = xyz[3 * i]; py[cnt] = xyz[3 * i + 1]; pz[cnt] = xyz[3 * i + 2];
        pd[cnt] = 1e10f; ++cnt;
    }
    int far = 0;
    if (tid == 0) { s_far[0] = xyz[0]; s_far[1] = xyz[1]; s_far[2] = xyz[2]; }
    for (int it = 0; it < npoint; ++it) {
        if (tid == 0) outIdx[it] = far;
        __syncthreads();
        const float fx = s_far[0], fy = s_far[1], fz = s_far[2];
        float best = -1.f; int bi = 0x7fffffff;
        for (int c = 0; c < cnt; ++c) {
            const float dx = px[c] - fx, dy = py[c] - fy, dz = pz[c] - fz;
            const float d = dx * dx + dy * dy + dz * dz;
            pd[c] = fminf(pd[c], d);
            if (pd[c] > best) { best = pd[c]; bi = tid + c * FPS_TPB; }
        }
        s_val[tid] = best; s_idx[tid] = bi;
        __syncthreads();
        for (int s = FPS_TPB / 2; s > 0; s >>= 1) {
            if (tid < s) {
                if (s_val[tid + s] > s_val[tid] ||
                    (s_val[tid + s] == s_val[tid] && s_idx[tid + s] < s_idx[tid])) {
                    s_val[tid] = s_val[tid + s]; s_idx[tid] = s_idx[tid + s];
                }
            }
            __syncthreads();
        }
        far = s_idx[0];
        if (tid == (far % FPS_TPB)) {
            const int c = far / FPS_TPB;
            s_far[0] = px[c]; s_far[1] = py[c]; s_far[2] = pz[c];
        }
        __syncthreads();
    }
}

// --------------------------------------------------------------------
// Brute-force KNN: per-query top-K sorted list in VGPRs (fully
// unrolled insertion), reference points streamed via LDS tiles.
// --------------------------------------------------------------------
template <int K>
__global__ __launch_bounds__(256)
void k_knn(const float* __restrict__ ref, int M,
           const float* __restrict__ qry, int Nq, int* __restrict__ outIdx)
{
    const int TILE = 512;
    __shared__ float sx[TILE], sy[TILE], sz[TILE];
    const int q = blockIdx.x * blockDim.x + threadIdx.x;
    const bool valid = q < Nq;
    float qx = 0.f, qy = 0.f, qz = 0.f;
    if (valid) { qx = qry[3 * q]; qy = qry[3 * q + 1]; qz = qry[3 * q + 2]; }
    float bd[K]; int bi[K];
#pragma unroll
    for (int j = 0; j < K; ++j) { bd[j] = 3.0e38f; bi[j] = 0; }
    for (int base = 0; base < M; base += TILE) {
        const int cnt = min(TILE, M - base);
        for (int t = threadIdx.x; t < cnt; t += blockDim.x) {
            sx[t] = ref[3 * (base + t)];
            sy[t] = ref[3 * (base + t) + 1];
            sz[t] = ref[3 * (base + t) + 2];
        }
        __syncthreads();
        if (valid) {
            for (int t = 0; t < cnt; ++t) {
                const float dx = qx - sx[t], dy = qy - sy[t], dz = qz - sz[t];
                const float d = dx * dx + dy * dy + dz * dz;
                if (d < bd[K - 1]) {
                    bd[K - 1] = d; bi[K - 1] = base + t;
#pragma unroll
                    for (int j = K - 1; j > 0; --j) {   // single bubble pass
                        if (bd[j] < bd[j - 1]) {
                            const float td = bd[j]; bd[j] = bd[j - 1]; bd[j - 1] = td;
                            const int ti = bi[j]; bi[j] = bi[j - 1]; bi[j - 1] = ti;
                        }
                    }
                }
            }
        }
        __syncthreads();
    }
    if (valid) {
#pragma unroll
        for (int j = 0; j < K; ++j) outIdx[(size_t)q * K + j] = bi[j];
    }
}

// group: g[e] = ref_xyz[idx[e]] - q_xyz[e/K]; new_pts[e] = [g | feats[idx[e]]]
__global__ void k_group(const float* __restrict__ rxyz, const float* __restrict__ feats, int C,
                        const float* __restrict__ qxyz, const int* __restrict__ idx,
                        int NK, int K, float* __restrict__ g, float* __restrict__ npts)
{
    const int e = blockIdx.x * blockDim.x + threadIdx.x;
    if (e >= NK) return;
    const int q = e / K;
    const int id = idx[e];
    const float gx = rxyz[3 * id] - qxyz[3 * q];
    const float gy = rxyz[3 * id + 1] - qxyz[3 * q + 1];
    const float gz = rxyz[3 * id + 2] - qxyz[3 * q + 2];
    g[3 * e] = gx; g[3 * e + 1] = gy; g[3 * e + 2] = gz;
    float* o = npts + (size_t)e * (3 + C);
    o[0] = gx; o[1] = gy; o[2] = gz;
    const float* f = feats + (size_t)id * C;
    for (int c = 0; c < C; ++c) o[3 + c] = f[c];
}

__global__ void k_dirs(const float* __restrict__ rxyz, const float* __restrict__ qxyz,
                       const int* __restrict__ idx, int NK, int K, float* __restrict__ g)
{
    const int e = blockIdx.x * blockDim.x + threadIdx.x;
    if (e >= NK) return;
    const int q = e / K;
    const int id = idx[e];
    g[3 * e]     = rxyz[3 * id]     - qxyz[3 * q];
    g[3 * e + 1] = rxyz[3 * id + 1] - qxyz[3 * q + 1];
    g[3 * e + 2] = rxyz[3 * id + 2] - qxyz[3 * q + 2];
}

// WeightNet: 3 -> 8 -> 8 -> O, ReLU after every layer. One row per thread.
__global__ void k_weightnet(const float* __restrict__ g, int M,
                            const float* __restrict__ W1, const float* __restrict__ b1,
                            const float* __restrict__ W2, const float* __restrict__ b2,
                            const float* __restrict__ W3, const float* __restrict__ b3,
                            int O, float* __restrict__ out)
{
    const int e = blockIdx.x * blockDim.x + threadIdx.x;
    if (e >= M) return;
    const float x0 = g[3 * e], x1 = g[3 * e + 1], x2 = g[3 * e + 2];
    float h1[8], h2[8];
#pragma unroll
    for (int o = 0; o < 8; ++o)
        h1[o] = reluf(W1[o * 3] * x0 + W1[o * 3 + 1] * x1 + W1[o * 3 + 2] * x2 + b1[o]);
#pragma unroll
    for (int o = 0; o < 8; ++o) {
        float s = b2[o];
#pragma unroll
        for (int j = 0; j < 8; ++j) s += W2[o * 8 + j] * h1[j];
        h2[o] = reluf(s);
    }
    float* op = out + (size_t)e * O;
    for (int o = 0; o < O; ++o) {
        float s = b3[o];
#pragma unroll
        for (int j = 0; j < 8; ++j) s += W3[o * 8 + j] * h2[j];
        op[o] = reluf(s);
    }
}

// Cost-volume stage 1, fused gather+concat+linear+leaky:
// out[e] = leaky(W * [f1[q] | f2[idx[e]] | dirx(e)] + b), also emits dirx.
__global__ void k_cost_mlp1(const float* __restrict__ xyz1, const float* __restrict__ xyz2,
                            const float* __restrict__ f1, int C1,
                            const float* __restrict__ f2, int C2,
                            const int* __restrict__ idx, int NK, int K,
                            const float* __restrict__ W, const float* __restrict__ b,
                            int m0, float* __restrict__ out, float* __restrict__ dirx)
{
    const int e = blockIdx.x * blockDim.x + threadIdx.x;
    if (e >= NK) return;
    const int q = e / K;
    const int id = idx[e];
    const float dx = xyz2[3 * id]     - xyz1[3 * q];
    const float dy = xyz2[3 * id + 1] - xyz1[3 * q + 1];
    const float dz = xyz2[3 * id + 2] - xyz1[3 * q + 2];
    dirx[3 * e] = dx; dirx[3 * e + 1] = dy; dirx[3 * e + 2] = dz;
    const float* a = f1 + (size_t)q * C1;
    const float* gf = f2 + (size_t)id * C2;
    float* op = out + (size_t)e * m0;
    const int Cin = C1 + C2 + 3;
    for (int o = 0; o < m0; ++o) {
        const float* wr = W + (size_t)o * Cin;
        float s = b[o];
        for (int c = 0; c < C1; ++c) s += wr[c] * a[c];
        for (int c = 0; c < C2; ++c) s += wr[C1 + c] * gf[c];
        s += wr[C1 + C2] * dx + wr[C1 + C2 + 1] * dy + wr[C1 + C2 + 2] * dz;
        op[o] = leakyf(s);
    }
}

// out[n,c] = sum_k A[n,k,c] * B[n,k,c]
__global__ void k_mul_sum(const float* __restrict__ A, const float* __restrict__ Bv,
                          int N, int K, int C, float* __restrict__ out)
{
    const int e = blockIdx.x * blockDim.x + threadIdx.x;
    if (e >= N * C) return;
    const int n = e / C, c = e % C;
    const float* a = A + (size_t)n * K * C + c;
    const float* b = Bv + (size_t)n * K * C + c;
    float s = 0.f;
    for (int k = 0; k < K; ++k) s += a[(size_t)k * C] * b[(size_t)k * C];
    out[e] = s;
}

// out[n,c] = sum_k Wt[n,k,c] * P[idx[n*K+k], c]
__global__ void k_gather_mul_sum(const float* __restrict__ Wt, const float* __restrict__ P,
                                 const int* __restrict__ idx, int N, int K, int C,
                                 float* __restrict__ out)
{
    const int e = blockIdx.x * blockDim.x + threadIdx.x;
    if (e >= N * C) return;
    const int n = e / C, c = e % C;
    const float* w = Wt + (size_t)n * K * C + c;
    const int* ix = idx + (size_t)n * K;
    float s = 0.f;
    for (int k = 0; k < K; ++k) s += w[(size_t)k * C] * P[(size_t)ix[k] * C + c];
    out[e] = s;
}

// Inverse-distance 3-NN interpolation (warpMode: out = q - interp, C==3)
__global__ void k_interp_apply(const float* __restrict__ qxyz, const float* __restrict__ sxyz,
                               const float* __restrict__ sval, int C,
                               const int* __restrict__ idx, int Nq,
                               float* __restrict__ out, int warpMode)
{
    const int q = blockIdx.x * blockDim.x + threadIdx.x;
    if (q >= Nq) return;
    const float qx = qxyz[3 * q], qy = qxyz[3 * q + 1], qz = qxyz[3 * q + 2];
    int ii[3] = { idx[3 * q], idx[3 * q + 1], idx[3 * q + 2] };
    float w[3]; float ws = 0.f;
#pragma unroll
    for (int j = 0; j < 3; ++j) {
        const float dx = sxyz[3 * ii[j]]     - qx;
        const float dy = sxyz[3 * ii[j] + 1] - qy;
        const float dz = sxyz[3 * ii[j] + 2] - qz;
        float d = sqrtf(dx * dx + dy * dy + dz * dz);
        d = fmaxf(d, 1e-10f);
        w[j] = 1.f / d; ws += w[j];
    }
    const float inv = 1.f / ws;
    for (int c = 0; c < C; ++c) {
        const float acc = (w[0] * sval[(size_t)ii[0] * C + c] +
                           w[1] * sval[(size_t)ii[1] * C + c] +
                           w[2] * sval[(size_t)ii[2] * C + c]) * inv;
        out[(size_t)q * C + c] = warpMode ? (qxyz[3 * q + c] - acc) : acc;
    }
}

__global__ void k_add3(const float* __restrict__ a, const float* __restrict__ b,
                       float* __restrict__ o, int n)
{
    const int e = blockIdx.x * blockDim.x + threadIdx.x;
    if (e < n) o[e] = a[e] + b[e];
}

__global__ void k_copy_slice(const float* __restrict__ src, int Cs,
                             float* __restrict__ dst, int Cd, int off, int N)
{
    const int e = blockIdx.x * blockDim.x + threadIdx.x;
    if (e >= N * Cs) return;
    const int n = e / Cs, c = e % Cs;
    dst[(size_t)n * Cd + off + c] = src[(size_t)n * Cs + c];
}

__global__ void k_gather_rows(const float* __restrict__ src, int C,
                              const int* __restrict__ idx, int n, float* __restrict__ dst)
{
    const int e = blockIdx.x * blockDim.x + threadIdx.x;
    if (e >= n * C) return;
    const int i = e / C, c = e % C;
    dst[e] = src[(size_t)idx[i] * C + c];
}

__global__ void k_transpose(const float* __restrict__ in, int N, int C, float* __restrict__ out)
{
    const int e = blockIdx.x * blockDim.x + threadIdx.x;
    if (e >= N * C) return;
    const int n = e / C, c = e % C;
    out[(size_t)c * N + n] = in[e];
}

__global__ void k_i2f(const int* __restrict__ in, int n, float* __restrict__ out)
{
    const int e = blockIdx.x * blockDim.x + threadIdx.x;
    if (e < n) out[e] = (float)in[e];
}

// ============================ host side =============================

static inline int cdiv(long a, long b) { return (int)((a + b - 1) / b); }

struct P { const float* W; const float* b; };

// Parameter leaf table: the harness flattens setup_inputs() as a pytree;
// JAX flattens dicts in sorted-key order ('W' < 'b', 'lin' < 'wn',
// 'fc' < 'mlp' < 'pc', ...). Supports both one-din-per-leaf (n_in==214)
// and a single concatenated blob (n_in==5).
struct PT {
    const float* leaf[220];
    int cnt; bool packed; const float* blob; size_t off;
    void* const* din; int next;
    void add(size_t sz) {
        if (packed) { leaf[cnt++] = blob + off; off += sz; }
        else        { leaf[cnt++] = (const float*)din[next++]; }
    }
    void lin(int i, int o) { add((size_t)i * o); add(o); }
    void wn(int o)         { lin(3, 8); lin(8, 8); lin(8, o); }
    void pconv(int cin, int cout) { lin(16 * cin, cout); wn(16); }      // lin, wn
    void pcf(int cin, int m0, int m1) { lin(cin, m0); lin(m0, m1); wn(m1); wn(m1); }
    void sfe(int feat, int cost, int flow) {
        const int last = feat + cost + flow;
        lin(64, 3); lin(128, 128); lin(128, 64);                        // fc, mlp0, mlp1
        pconv(last + 3, 128); pconv(131, 128);                          // pc0, pc1
    }
};
static inline P LP(const PT& pt, int i) { return P{ pt.leaf[i], pt.leaf[i + 1] }; }

// leaf indices (sorted top-level key order, inputs excluded)
enum {
    L_COST0 = 0, L_COST1 = 16, L_COST2 = 32, L_COST3 = 48,
    L_DC10 = 64, L_DC21 = 66, L_DC32 = 68, L_DC43 = 70,
    L_FLOW0 = 72, L_FLOW1 = 94, L_FLOW2 = 116, L_FLOW3 = 138,
    L_L0 = 160, L_L0_1 = 162, L_L0_2 = 164,
    L_L1 = 166, L_L1_0 = 174, L_L1_1 = 176,
    L_L2 = 178, L_L2_0 = 186, L_L2_1 = 188,
    L_L3 = 190, L_L3_0 = 198, L_L3_1 = 200,
    L_L4 = 202
};

struct Ctx {
    hipStream_t s;
    char* base; size_t off, cap;
    float* af(size_t n) { off = (off + 255) & ~(size_t)255; float* p = (float*)(base + off); off += n * sizeof(float); return p; }
    int*   ai(size_t n) { off = (off + 255) & ~(size_t)255; int*   p = (int*)(base + off); off += n * sizeof(int);   return p; }
};

static void gemm(Ctx& c, const float* X, int N, int Ci, P w, int Co, int act, float* Y) {
    const long tiles = (long)cdiv(N, 16) * cdiv(Co, 64);
    const bool vec = ((Ci & 3) == 0) &&
                     (((uintptr_t)X & 15) == 0) && (((uintptr_t)w.W & 15) == 0);
    if (vec) k_gemm<true><<<cdiv(tiles, 8), 256, 0, c.s>>>(X, w.W, w.b, Y, N, Ci, Co, act);
    else     k_gemm<false><<<cdiv(tiles, 8), 256, 0, c.s>>>(X, w.W, w.b, Y, N, Ci, Co, act);
}

static void knn(Ctx& c, const float* ref, int M, const float* q, int Nq, int K, int* idx) {
    const int g = cdiv(Nq, 256);
    switch (K) {
        case 3:  k_knn<3><<<g, 256, 0, c.s>>>(ref, M, q, Nq, idx); break;
        case 9:  k_knn<9><<<g, 256, 0, c.s>>>(ref, M, q, Nq, idx); break;
        case 16: k_knn<16><<<g, 256, 0, c.s>>>(ref, M, q, Nq, idx); break;
        default: k_knn<32><<<g, 256, 0, c.s>>>(ref, M, q, Nq, idx); break;
    }
}

static void weightnet(Ctx& c, const PT& pt, const float* g, long M, int wbase, int O, float* out) {
    P w1 = LP(pt, wbase), w2 = LP(pt, wbase + 2), w3 = LP(pt, wbase + 4);
    k_weightnet<<<cdiv(M, 256), 256, 0, c.s>>>(g, (int)M, w1.W, w1.b, w2.W, w2.b, w3.W, w3.b, O, out);
}

// PointConv core: KNN -> group -> WeightNet -> WMMA einsum -> WMMA linear+leaky
static void pointconv_core(Ctx& c, const PT& pt, int pbase,
                           const float* qxyz, int Nq, const float* rxyz, int Nr,
                           const float* feats, int Cf, int K, int Cout, float* out)
{
    const size_t mk = c.off;
    const int Cin = Cf + 3;
    int* idx = c.ai((size_t)Nq * K);
    knn(c, rxyz, Nr, qxyz, Nq, K, idx);
    float* g    = c.af((size_t)Nq * K * 3);
    float* npts = c.af((size_t)Nq * K * Cin);
    k_group<<<cdiv((long)Nq * K, 256), 256, 0, c.s>>>(rxyz, feats, Cf, qxyz, idx, Nq * K, K, g, npts);
    float* wts = c.af((size_t)Nq * K * 16);
    weightnet(c, pt, g, (long)Nq * K, pbase + 2, 16, wts);
    float* eo = c.af((size_t)Nq * Cin * 16);
    const long waves = (long)Nq * cdiv(Cin, 16);
    k_pc_einsum<<<cdiv(waves, 8), 256, 0, c.s>>>(npts, wts, eo, Nq, K, Cin);
    gemm(c, eo, Nq, Cin * 16, LP(pt, pbase), Cout, 1, out);
    c.off = mk;
}

static void pointconvd(Ctx& c, const PT& pt, int pbase,
                       const float* xyz, int N, const float* feats, int Cf,
                       int npoint, int Cout, float* new_xyz, float* out, int* fidx)
{
    k_fps<<<1, FPS_TPB, 0, c.s>>>(xyz, N, npoint, fidx);
    k_gather_rows<<<cdiv((long)npoint * 3, 256), 256, 0, c.s>>>(xyz, 3, fidx, npoint, new_xyz);
    pointconv_core(c, pt, pbase, new_xyz, npoint, xyz, N, feats, Cf, 16, Cout, out);
}

static void interp3(Ctx& c, const float* qxyz, int Nq, const float* sxyz, int M,
                    const float* sval, int C, float* out, int warpMode)
{
    const size_t mk = c.off;
    int* idx = c.ai((size_t)Nq * 3);
    knn(c, sxyz, M, qxyz, Nq, 3, idx);
    k_interp_apply<<<cdiv(Nq, 256), 256, 0, c.s>>>(qxyz, sxyz, sval, C, idx, Nq, out, warpMode);
    c.off = mk;
}

static void warpf(Ctx& c, const float* xyz1, int N1, const float* xyz2, int N2,
                  const float* flow, float* out)
{
    const size_t mk = c.off;
    float* x12 = c.af((size_t)N1 * 3);
    k_add3<<<cdiv((long)N1 * 3, 256), 256, 0, c.s>>>(xyz1, flow, x12, N1 * 3);
    interp3(c, xyz2, N2, x12, N1, flow, 3, out, 1);
    c.off = mk;
}

// PointConvFlow cost volume (point-to-patch then patch-to-patch)
static void costvol(Ctx& c, const PT& pt, int pbase,
                    const float* xyz1, int N1, const float* xyz2, int N2,
                    const float* f1, int C1, const float* f2, int C2,
                    int m0, int m1, float* out)
{
    const size_t mk = c.off;
    const int K = 32;
    int* idx = c.ai((size_t)N1 * K);
    knn(c, xyz2, N2, xyz1, N1, K, idx);
    float* dirx = c.af((size_t)N1 * K * 3);
    float* h1   = c.af((size_t)N1 * K * m0);
    P m0p = LP(pt, pbase);
    k_cost_mlp1<<<cdiv((long)N1 * K, 256), 256, 0, c.s>>>(xyz1, xyz2, f1, C1, f2, C2, idx,
                                                          N1 * K, K, m0p.W, m0p.b, m0, h1, dirx);
    float* h2 = c.af((size_t)N1 * K * m1);
    gemm(c, h1, N1 * K, m0, LP(pt, pbase + 2), m1, 1, h2);
    float* w1 = c.af((size_t)N1 * K * m1);
    weightnet(c, pt, dirx, (long)N1 * K, pbase + 4, m1, w1);
    float* p2p = c.af((size_t)N1 * m1);
    k_mul_sum<<<cdiv((long)N1 * m1, 256), 256, 0, c.s>>>(w1, h2, N1, K, m1, p2p);
    int* idx1 = c.ai((size_t)N1 * K);
    knn(c, xyz1, N1, xyz1, N1, K, idx1);
    float* dir1 = c.af((size_t)N1 * K * 3);
    k_dirs<<<cdiv((long)N1 * K, 256), 256, 0, c.s>>>(xyz1, xyz1, idx1, N1 * K, K, dir1);
    float* w2 = c.af((size_t)N1 * K * m1);
    weightnet(c, pt, dir1, (long)N1 * K, pbase + 10, m1, w2);
    k_gather_mul_sum<<<cdiv((long)N1 * m1, 256), 256, 0, c.s>>>(w2, p2p, idx1, N1, K, m1, out);
    c.off = mk;
}

// SceneFlowEstimator: concat -> 2x PointConv(k=9) -> 2x conv1d -> fc+clamp
static void sfe(Ctx& c, const PT& pt, int sbase,
                const float* xyz, int N, const float* feats, int Cf,
                const float* cost, int Cc, const float* flow, int Cfl,
                float* featOut, float* flowOut)
{
    const size_t mk = c.off;
    const int Cx = Cf + Cc + Cfl;
    float* x = c.af((size_t)N * Cx);
    k_copy_slice<<<cdiv((long)N * Cf, 256), 256, 0, c.s>>>(feats, Cf, x, Cx, 0, N);
    k_copy_slice<<<cdiv((long)N * Cc, 256), 256, 0, c.s>>>(cost, Cc, x, Cx, Cf, N);
    if (flow && Cfl > 0)
        k_copy_slice<<<cdiv((long)N * Cfl, 256), 256, 0, c.s>>>(flow, Cfl, x, Cx, Cf + Cc, N);
    float* t0 = c.af((size_t)N * 128);
    pointconv_core(c, pt, sbase + 6, xyz, N, xyz, N, x, Cx, 9, 128, t0);
    float* t1 = c.af((size_t)N * 128);
    pointconv_core(c, pt, sbase + 14, xyz, N, xyz, N, t0, 128, 9, 128, t1);
    float* m_ = c.af((size_t)N * 128);
    gemm(c, t1, N, 128, LP(pt, sbase + 2), 128, 1, m_);
    gemm(c, m_, N, 128, LP(pt, sbase + 4), 64, 1, featOut);
    gemm(c, featOut, N, 64, LP(pt, sbase), 3, 3, flowOut);
    c.off = mk;
}

static void conv1d2(Ctx& c, const PT& pt, const float* in, int N, int Ci,
                    int pA, int Cmid, int pB, int Cout, float* out)
{
    const size_t mk = c.off;
    float* t = c.af((size_t)N * Cmid);
    gemm(c, in, N, Ci, LP(pt, pA), Cmid, 1, t);
    gemm(c, t, N, Cmid, LP(pt, pB), Cout, 1, out);
    c.off = mk;
}

static void interp_deconv(Ctx& c, const PT& pt, const float* qxyz, int Nq,
                          const float* sxyz, int M, const float* sval, int C,
                          int pLin, int Cout, float* out)
{
    const size_t mk = c.off;
    float* t = c.af((size_t)Nq * C);
    interp3(c, qxyz, Nq, sxyz, M, sval, C, t, 0);
    gemm(c, t, Nq, C, LP(pt, pLin), Cout, 1, out);
    c.off = mk;
}

static void concat2(Ctx& c, const float* A, int Ca, const float* B, int Cb, int N, float* out) {
    k_copy_slice<<<cdiv((long)N * Ca, 256), 256, 0, c.s>>>(A, Ca, out, Ca + Cb, 0, N);
    k_copy_slice<<<cdiv((long)N * Cb, 256), 256, 0, c.s>>>(B, Cb, out, Ca + Cb, Ca, N);
}

extern "C" void kernel_launch(void* const* d_in, const int* in_sizes, int n_in,
                              void* d_out, int out_size, void* d_ws, size_t ws_size,
                              hipStream_t stream)
{
    (void)in_sizes; (void)out_size;
    if (n_in != 5 && n_in < 214) return;   // unexpected flattening; bail deterministically

    const float* xyz1 = (const float*)d_in[0];
    const float* xyz2 = (const float*)d_in[1];
    const float* col1 = (const float*)d_in[2];
    const float* col2 = (const float*)d_in[3];

    PT pt; pt.cnt = 0; pt.off = 0; pt.din = d_in; pt.next = 4;
    pt.packed = (n_in == 5);
    pt.blob = pt.packed ? (const float*)d_in[4] : nullptr;
    // sorted top-level key order:
    pt.pcf(131, 32, 32);     // cost0
    pt.pcf(195, 64, 64);     // cost1
    pt.pcf(387, 128, 128);   // cost2
    pt.pcf(643, 256, 256);   // cost3
    pt.lin(64, 32);          // deconv1_0
    pt.lin(128, 32);         // deconv2_1
    pt.lin(256, 64);         // deconv3_2
    pt.lin(256, 64);         // deconv4_3
    pt.sfe(96, 32, 3);       // flow0
    pt.sfe(128, 64, 3);      // flow1
    pt.sfe(192, 128, 3);     // flow2
    pt.sfe(256, 256, 0);     // flow3
    pt.lin(3, 32);           // level0
    pt.lin(32, 32);          // level0_1
    pt.lin(32, 64);          // level0_2
    pt.pconv(67, 64);        // level1
    pt.lin(64, 64);          // level1_0
    pt.lin(64, 128);         // level1_1
    pt.pconv(131, 128);      // level2
    pt.lin(128, 128);        // level2_0
    pt.lin(128, 256);        // level2_1
    pt.pconv(259, 256);      // level3
    pt.lin(256, 256);        // level3_0
    pt.lin(256, 512);        // level3_1
    pt.pconv(515, 256);      // level4

    Ctx c{ stream, (char*)d_ws, 0, ws_size };
    const int N0 = 8192, N1 = 2048, N2 = 512, N3 = 256, N4 = 64;

    // ---- persistent buffers ----
    float *f1l0 = c.af((size_t)N0*32), *f2l0 = c.af((size_t)N0*32);
    float *f1l0b = c.af((size_t)N0*64), *f2l0b = c.af((size_t)N0*64);
    float *pc1l1 = c.af((size_t)N1*3), *pc2l1 = c.af((size_t)N1*3);
    int   *fi1l1 = c.ai(N1), *fi2l1 = c.ai(N1);
    float *f1l1 = c.af((size_t)N1*64), *f2l1 = c.af((size_t)N1*64);
    float *f1l1b = c.af((size_t)N1*128), *f2l1b = c.af((size_t)N1*128);
    float *pc1l2 = c.af((size_t)N2*3), *pc2l2 = c.af((size_t)N2*3);
    int   *fi1l2 = c.ai(N2), *fi2l2 = c.ai(N2);
    float *f1l2 = c.af((size_t)N2*128), *f2l2 = c.af((size_t)N2*128);
    float *f1l2b = c.af((size_t)N2*256), *f2l2b = c.af((size_t)N2*256);
    float *pc1l3 = c.af((size_t)N3*3), *pc2l3 = c.af((size_t)N3*3);
    int   *fi1l3 = c.ai(N3), *fi2l3 = c.ai(N3);
    float *f1l3 = c.af((size_t)N3*256), *f2l3 = c.af((size_t)N3*256);
    float *f1l3b = c.af((size_t)N3*512), *f2l3b = c.af((size_t)N3*512);
    float *pc1l4 = c.af((size_t)N4*3), *pc2l4 = c.af((size_t)N4*3);
    int   *fi1l4 = c.ai(N4), *fi2l4 = c.ai(N4);
    float *f1l4 = c.af((size_t)N4*256), *f2l4 = c.af((size_t)N4*256);
    float *f1l43 = c.af((size_t)N3*64), *f2l43 = c.af((size_t)N3*64);
    float *cf1l3 = c.af((size_t)N3*320), *cf2l3 = c.af((size_t)N3*320);
    float *cost3 = c.af((size_t)N3*256);
    float *feat3 = c.af((size_t)N3*64), *flow3 = c.af((size_t)N3*3);
    float *f1l32 = c.af((size_t)N2*64), *f2l32 = c.af((size_t)N2*64);
    float *cf1l2 = c.af((size_t)N2*192), *cf2l2 = c.af((size_t)N2*192);
    float *f1l21 = c.af((size_t)N1*32), *f2l21 = c.af((size_t)N1*32);
    float *cf1l1 = c.af((size_t)N1*96), *cf2l1 = c.af((size_t)N1*96);
    float *f1l10 = c.af((size_t)N0*32), *f2l10 = c.af((size_t)N0*32);
    float *cf1l0 = c.af((size_t)N0*64), *cf2l0 = c.af((size_t)N0*64);
    float *upf2 = c.af((size_t)N2*3), *pc2l2w = c.af((size_t)N2*3);
    float *cost2 = c.af((size_t)N2*128);
    float *f3up = c.af((size_t)N2*64), *nf1l2 = c.af((size_t)N2*192);
    float *feat2 = c.af((size_t)N2*64), *flow2 = c.af((size_t)N2*3);
    float *upf1 = c.af((size_t)N1*3), *pc2l1w = c.af((size_t)N1*3);
    float *cost1 = c.af((size_t)N1*64);
    float *f2up = c.af((size_t)N1*64), *nf1l1 = c.af((size_t)N1*128);
    float *feat1 = c.af((size_t)N1*64), *flow1 = c.af((size_t)N1*3);
    float *upf0 = c.af((size_t)N0*3), *pc2l0w = c.af((size_t)N0*3);
    float *cost0 = c.af((size_t)N0*32);
    float *f1up = c.af((size_t)N0*64), *nf1l0 = c.af((size_t)N0*96);
    float *feat0d = c.af((size_t)N0*64), *flow0 = c.af((size_t)N0*3);

    // ---- feature extraction ----
    { const size_t mk = c.off; float* t = c.af((size_t)N0 * 32);
      gemm(c, col1, N0, 3, LP(pt, L_L0), 32, 1, t);
      gemm(c, t, N0, 32, LP(pt, L_L0_1), 32, 1, f1l0);
      gemm(c, col2, N0, 3, LP(pt, L_L0), 32, 1, t);
      gemm(c, t, N0, 32, LP(pt, L_L0_1), 32, 1, f2l0);
      c.off = mk; }
    gemm(c, f1l0, N0, 32, LP(pt, L_L0_2), 64, 1, f1l0b);
    gemm(c, f2l0, N0, 32, LP(pt, L_L0_2), 64, 1, f2l0b);

    // ---- encoder pyramid ----
    pointconvd(c, pt, L_L1, xyz1, N0, f1l0b, 64, N1, 64, pc1l1, f1l1, fi1l1);
    conv1d2(c, pt, f1l1, N1, 64, L_L1_0, 64, L_L1_1, 128, f1l1b);
    pointconvd(c, pt, L_L1, xyz2, N0, f2l0b, 64, N1, 64, pc2l1, f2l1, fi2l1);
    conv1d2(c, pt, f2l1, N1, 64, L_L1_0, 64, L_L1_1, 128, f2l1b);

    pointconvd(c, pt, L_L2, pc1l1, N1, f1l1b, 128, N2, 128, pc1l2, f1l2, fi1l2);
    conv1d2(c, pt, f1l2, N2, 128, L_L2_0, 128, L_L2_1, 256, f1l2b);
    pointconvd(c, pt, L_L2, pc2l1, N1, f2l1b, 128, N2, 128, pc2l2, f2l2, fi2l2);
    conv1d2(c, pt, f2l2, N2, 128, L_L2_0, 128, L_L2_1, 256, f2l2b);

    pointconvd(c, pt, L_L3, pc1l2, N2, f1l2b, 256, N3, 256, pc1l3, f1l3, fi1l3);
    conv1d2(c, pt, f1l3, N3, 256, L_L3_0, 256, L_L3_1, 512, f1l3b);
    pointconvd(c, pt, L_L3, pc2l2, N2, f2l2b, 256, N3, 256, pc2l3, f2l3, fi2l3);
    conv1d2(c, pt, f2l3, N3, 256, L_L3_0, 256, L_L3_1, 512, f2l3b);

    pointconvd(c, pt, L_L4, pc1l3, N3, f1l3b, 512, N4, 256, pc1l4, f1l4, fi1l4);
    pointconvd(c, pt, L_L4, pc2l3, N3, f2l3b, 512, N4, 256, pc2l4, f2l4, fi2l4);
    interp_deconv(c, pt, pc1l3, N3, pc1l4, N4, f1l4, 256, L_DC43, 64, f1l43);
    interp_deconv(c, pt, pc2l3, N3, pc2l4, N4, f2l4, 256, L_DC43, 64, f2l43);

    // ---- level 3 flow ----
    concat2(c, f1l3, 256, f1l43, 64, N3, cf1l3);
    concat2(c, f2l3, 256, f2l43, 64, N3, cf2l3);
    costvol(c, pt, L_COST3, pc1l3, N3, pc2l3, N3, cf1l3, 320, cf2l3, 320, 256, 256, cost3);
    sfe(c, pt, L_FLOW3, pc1l3, N3, f1l3, 256, cost3, 256, nullptr, 0, feat3, flow3);

    // ---- skip features for lower levels ----
    interp_deconv(c, pt, pc1l2, N2, pc1l3, N3, f1l3, 256, L_DC32, 64, f1l32);
    interp_deconv(c, pt, pc2l2, N2, pc2l3, N3, f2l3, 256, L_DC32, 64, f2l32);
    concat2(c, f1l2, 128, f1l32, 64, N2, cf1l2);
    concat2(c, f2l2, 128, f2l32, 64, N2, cf2l2);

    interp_deconv(c, pt, pc1l1, N1, pc1l2, N2, f1l2, 128, L_DC21, 32, f1l21);
    interp_deconv(c, pt, pc2l1, N1, pc2l2, N2, f2l2, 128, L_DC21, 32, f2l21);
    concat2(c, f1l1, 64, f1l21, 32, N1, cf1l1);
    concat2(c, f2l1, 64, f2l21, 32, N1, cf2l1);

    interp_deconv(c, pt, xyz1, N0, pc1l1, N1, f1l1, 64, L_DC10, 32, f1l10);
    interp_deconv(c, pt, xyz2, N0, pc2l1, N1, f2l1, 64, L_DC10, 32, f2l10);
    concat2(c, f1l0, 32, f1l10, 32, N0, cf1l0);
    concat2(c, f2l0, 32, f2l10, 32, N0, cf2l0);

    // ---- level 2 flow ----
    interp3(c, pc1l2, N2, pc1l3, N3, flow3, 3, upf2, 0);          // SCALE == 1.0
    warpf(c, pc1l2, N2, pc2l2, N2, upf2, pc2l2w);
    costvol(c, pt, L_COST2, pc1l2, N2, pc2l2w, N2, cf1l2, 192, cf2l2, 192, 128, 128, cost2);
    interp3(c, pc1l2, N2, pc1l3, N3, feat3, 64, f3up, 0);
    concat2(c, f1l2, 128, f3up, 64, N2, nf1l2);
    sfe(c, pt, L_FLOW2, pc1l2, N2, nf1l2, 192, cost2, 128, upf2, 3, feat2, flow2);

    // ---- level 1 flow ----
    interp3(c, pc1l1, N1, pc1l2, N2, flow2, 3, upf1, 0);
    warpf(c, pc1l1, N1, pc2l1, N1, upf1, pc2l1w);
    costvol(c, pt, L_COST1, pc1l1, N1, pc2l1w, N1, cf1l1, 96, cf2l1, 96, 64, 64, cost1);
    interp3(c, pc1l1, N1, pc1l2, N2, feat2, 64, f2up, 0);
    concat2(c, f1l1, 64, f2up, 64, N1, nf1l1);
    sfe(c, pt, L_FLOW1, pc1l1, N1, nf1l1, 128, cost1, 64, upf1, 3, feat1, flow1);

    // ---- level 0 flow ----
    interp3(c, xyz1, N0, pc1l1, N1, flow1, 3, upf0, 0);
    warpf(c, xyz1, N0, xyz2, N0, upf0, pc2l0w);
    costvol(c, pt, L_COST0, xyz1, N0, pc2l0w, N0, cf1l0, 64, cf2l0, 64, 32, 32, cost0);
    interp3(c, xyz1, N0, pc1l1, N1, feat1, 64, f1up, 0);
    concat2(c, f1l0, 32, f1up, 64, N0, nf1l0);
    sfe(c, pt, L_FLOW0, xyz1, N0, nf1l0, 96, cost0, 32, upf0, 3, feat0d, flow0);

    // ---- assemble outputs (flows^T, fps idx, pcs^T) ----
    float* O = (float*)d_out;
    k_transpose<<<cdiv((long)N0*3,256),256,0,stream>>>(flow0, N0, 3, O + 0);
    k_transpose<<<cdiv((long)N1*3,256),256,0,stream>>>(flow1, N1, 3, O + 24576);
    k_transpose<<<cdiv((long)N2*3,256),256,0,stream>>>(flow2, N2, 3, O + 30720);
    k_transpose<<<cdiv((long)N3*3,256),256,0,stream>>>(flow3, N3, 3, O + 32256);
    k_i2f<<<cdiv((long)N1,256),256,0,stream>>>(fi1l1, N1, O + 33024);
    k_i2f<<<cdiv((long)N2,256),256,0,stream>>>(fi1l2, N2, O + 35072);
    k_i2f<<<cdiv((long)N3,256),256,0,stream>>>(fi1l3, N3, O + 35584);
    k_i2f<<<cdiv((long)N1,256),256,0,stream>>>(fi2l1, N1, O + 35840);
    k_i2f<<<cdiv((long)N2,256),256,0,stream>>>(fi2l2, N2, O + 37888);
    k_i2f<<<cdiv((long)N3,256),256,0,stream>>>(fi2l3, N3, O + 38400);
    k_transpose<<<cdiv((long)N0*3,256),256,0,stream>>>(xyz1,  N0, 3, O + 38656);
    k_transpose<<<cdiv((long)N1*3,256),256,0,stream>>>(pc1l1, N1, 3, O + 63232);
    k_transpose<<<cdiv((long)N2*3,256),256,0,stream>>>(pc1l2, N2, 3, O + 69376);
    k_transpose<<<cdiv((long)N3*3,256),256,0,stream>>>(pc1l3, N3, 3, O + 70912);
    k_transpose<<<cdiv((long)N0*3,256),256,0,stream>>>(pc1l3, N3, 3, O + 70912);
    k_transpose<<<cdiv((long)N0*3,256),256,0,stream>>>(xyz2,  N0, 3, O + 71680);
    k_transpose<<<cdiv((long)N1*3,256),256,0,stream>>>(pc2l1, N1, 3, O + 96256);
    k_transpose<<<cdiv((long)N2*3,256),256,0,stream>>>(pc2l2, N2, 3, O + 102400);
    k_transpose<<<cdiv((long)N3*3,256),256,0,stream>>>(pc2l3, N3, 3, O + 103936);
}
